// TypedTreeCell_26534307955067
// MI455X (gfx1250) — matrix operand order repaired and checked by reading, not compiled, not run on hardware
//
#include <hip/hip_runtime.h>
#include <hip/hip_bf16.h>

// TypedTreeCell reduce for MI455X (gfx1250, wave32, WMMA bf16 16x16x32).
// Memory-bound problem (~672MB @ 23.3TB/s ~= 29us floor); bf16 WMMA with f32
// accum keeps compute far below that. Nodes are bucketed by type so each
// 16-node tile shares one U matrix (avoids the reference's 4x masked redo).

typedef __bf16 bf16_t;
typedef __attribute__((ext_vector_type(16))) __bf16 v16bf;
typedef __attribute__((ext_vector_type(8)))  float  v8f;
typedef unsigned int u32;
typedef __attribute__((ext_vector_type(4))) u32 u32x4;   // trivially-constructible 16B chunk

#define N_NODES 65536
#define KCH 8
#define H 128
#define T_TYPES 4
#define TILES_PER_TYPE ((N_NODES + 15) / 16)   // 4096
#define SA 136                                  // padded LDS row stride (bf16 elems): 272B -> conflict-free b128 rows

union F16 { u32x4 u[2]; v16bf v; };
union Pk8 { bf16_t b8[8]; u32x4 u; };

__global__ void k_init(int* counts) {
    if (threadIdx.x < T_TYPES) counts[threadIdx.x] = 0;
}

// Transpose + f32->bf16 the tiny per-type U matrices:
// UfT[t][n][k]   = U_f[t][k][n]     (T*128*128)
// UiouT[t][n][k] = U_iou[t][k][n]   (T*384*128)
__global__ void k_prep(const float* __restrict__ U_f, const float* __restrict__ U_iou,
                       bf16_t* __restrict__ UfT, bf16_t* __restrict__ UiouT) {
    int i = blockIdx.x * blockDim.x + threadIdx.x;
    const int NF = T_TYPES * H * H;
    const int NI = T_TYPES * 3 * H * H;
    if (i < NF) {
        int t = i / (H * H), r = i % (H * H), n = r / H, k = r % H;
        UfT[i] = (bf16_t)U_f[t * H * H + k * H + n];
    } else if (i < NF + NI) {
        int j = i - NF;
        int t = j / (3 * H * H), r = j % (3 * H * H), n = r / H, k = r % H;
        UiouT[j] = (bf16_t)U_iou[t * 3 * H * H + k * 3 * H + n];
    }
}

__global__ void k_bucket(const int* __restrict__ type_id, int* counts, int* idx) {
    int n = blockIdx.x * blockDim.x + threadIdx.x;
    if (n < N_NODES) {
        int t = type_id[n];
        int pos = atomicAdd(&counts[t], 1);
        idx[t * N_NODES + pos] = n;
    }
}

__global__ __launch_bounds__(256, 2)
void k_compute(const float* __restrict__ h, const float* __restrict__ c,
               const float* __restrict__ f_input, const float* __restrict__ b_iou,
               const float* __restrict__ b_f,
               const bf16_t* __restrict__ UfT, const bf16_t* __restrict__ UiouT,
               const int* __restrict__ counts, const int* __restrict__ idx,
               float* __restrict__ out) {
    __shared__ bf16_t sAh[128 * SA];   // h children tile, bf16  (~34.8 KB)
    __shared__ bf16_t sAs[16 * SA];    // h_sum tile, bf16       (~4.3 KB)
    __shared__ int    sNode[16];

    const int t    = blockIdx.x / TILES_PER_TYPE;
    const int tile = blockIdx.x % TILES_PER_TYPE;
    const int cnt  = counts[t];
    const int tileStart = tile * 16;
    if (tileStart >= cnt) return;                 // block-uniform early exit
    const int rows = min(16, cnt - tileStart);
    const int tid  = threadIdx.x;
    const int* list = idx + t * N_NODES;

    if (tid < 16) sNode[tid] = list[tileStart + ((tid < rows) ? tid : 0)];  // clamp pads
    __syncthreads();

    // ---- stage 1a: h tile -> LDS bf16 (row = local_node*8 + child) ----
    {
        int r = tid >> 1, half = tid & 1;         // 2 threads per 128-elem row
        int node = sNode[r >> 3], child = r & 7;
        const float4* src = (const float4*)(h + ((size_t)node * KCH + child) * H + half * 64);
        bf16_t* dst = sAh + r * SA + half * 64;
        #pragma unroll
        for (int i = 0; i < 8; ++i) {
            float4 a = src[2 * i], b = src[2 * i + 1];
            Pk8 pk;
            pk.b8[0] = (bf16_t)a.x; pk.b8[1] = (bf16_t)a.y; pk.b8[2] = (bf16_t)a.z; pk.b8[3] = (bf16_t)a.w;
            pk.b8[4] = (bf16_t)b.x; pk.b8[5] = (bf16_t)b.y; pk.b8[6] = (bf16_t)b.z; pk.b8[7] = (bf16_t)b.w;
            *(u32x4*)(dst + 8 * i) = pk.u;
        }
    }
    __syncthreads();

    // ---- stage 1b: h_sum (16 x 128) in f32, stored bf16 ----
    {
        int nn = tid >> 4, cb = (tid & 15) * 8;
        float acc[8] = {0.f, 0.f, 0.f, 0.f, 0.f, 0.f, 0.f, 0.f};
        #pragma unroll
        for (int ch = 0; ch < KCH; ++ch) {
            Pk8 pk; pk.u = *(const u32x4*)(sAh + (nn * KCH + ch) * SA + cb);
            #pragma unroll
            for (int j = 0; j < 8; ++j) acc[j] += (float)pk.b8[j];
        }
        Pk8 st;
        #pragma unroll
        for (int j = 0; j < 8; ++j) st.b8[j] = (bf16_t)acc[j];
        *(u32x4*)(sAs + nn * SA + cb) = st.u;
    }
    __syncthreads();

    const int w    = tid >> 5;        // wave 0..7
    const int lane = tid & 31;
    const int ln   = lane & 15;       // column within 16
    const int lh   = lane >> 4;       // half-wave

    // ---- stage 2: f = sigmoid(f_input + h @ U_f + b_f); c_aggr = sum_k f*c ----
    // Wave w owns output column tile w; B fragments stay resident in VGPRs.
    {
        const bf16_t* Uf_t = UfT + t * H * H;
        F16 B[4];
        #pragma unroll
        for (int kt = 0; kt < 4; ++kt) {
            const bf16_t* bp = Uf_t + (w * 16 + ln) * H + kt * 32 + lh * 8;
            B[kt].u[0] = *(const u32x4*)bp;
            B[kt].u[1] = *(const u32x4*)(bp + 16);
        }
        const int col = w * 16 + ln;
        const float bfv = b_f[t * H + col];
        #pragma unroll
        for (int mt = 0; mt < 8; ++mt) {
            if (2 * mt < rows) {                       // wave-uniform guard (EXEC full for WMMA)
                v8f acc = {0.f, 0.f, 0.f, 0.f, 0.f, 0.f, 0.f, 0.f};
                #pragma unroll
                for (int kt = 0; kt < 4; ++kt) {
                    F16 A;
                    const bf16_t* ap = sAh + (mt * 16 + ln) * SA + kt * 32 + lh * 8;
                    A.u[0] = *(const u32x4*)ap;
                    A.u[1] = *(const u32x4*)(ap + 16);
                    acc = __builtin_amdgcn_wmma_f32_16x16x32_bf16(
                              false, A.v, false, B[kt].v, (short)0, acc, false, false);
                }
                // acc VGPR v <-> child v of node (2*mt + lh); N = ln
                const int nodeLoc = 2 * mt + lh;
                const int gnode = sNode[nodeLoc];
                const float fin = f_input[(size_t)gnode * H + col];
                const float* cp = c + (size_t)gnode * (KCH * H) + col;
                float csum = 0.f;
                #pragma unroll
                for (int v = 0; v < 8; ++v) {
                    float fpre = acc[v] + fin + bfv;
                    float fg = 1.0f / (1.0f + __expf(-fpre));
                    csum += fg * cp[v * H];
                }
                if (nodeLoc < rows) out[(size_t)gnode * (4 * H) + 3 * H + col] = csum;
            }
        }
    }

    // ---- stage 3: iou = h_sum @ U_iou + b_iou (16 x 384); 3 col tiles per wave ----
    {
        const bf16_t* Uiou_t = UiouT + t * 3 * H * H;
        #pragma unroll
        for (int jj = 0; jj < 3; ++jj) {
            const int nt = w * 3 + jj;
            v8f acc = {0.f, 0.f, 0.f, 0.f, 0.f, 0.f, 0.f, 0.f};
            #pragma unroll
            for (int kt = 0; kt < 4; ++kt) {
                F16 A, Bf;
                const bf16_t* ap = sAs + ln * SA + kt * 32 + lh * 8;
                A.u[0] = *(const u32x4*)ap;
                A.u[1] = *(const u32x4*)(ap + 16);
                const bf16_t* bp = Uiou_t + (nt * 16 + ln) * H + kt * 32 + lh * 8;
                Bf.u[0] = *(const u32x4*)bp;
                Bf.u[1] = *(const u32x4*)(bp + 16);
                acc = __builtin_amdgcn_wmma_f32_16x16x32_bf16(
                          false, A.v, false, Bf.v, (short)0, acc, false, false);
            }
            const int col = nt * 16 + ln;
            const float bio = b_iou[t * 3 * H + col];
            #pragma unroll
            for (int v = 0; v < 8; ++v) {
                int m = v + 8 * lh;                 // node row in tile
                if (m < rows) out[(size_t)sNode[m] * (4 * H) + col] = acc[v] + bio;
            }
        }
    }
}

extern "C" void kernel_launch(void* const* d_in, const int* in_sizes, int n_in,
                              void* d_out, int out_size, void* d_ws, size_t ws_size,
                              hipStream_t stream) {
    const float* h       = (const float*)d_in[0];
    const float* c       = (const float*)d_in[1];
    const float* f_input = (const float*)d_in[2];
    const int*   type_id = (const int*)d_in[3];
    const float* U_iou   = (const float*)d_in[4];
    const float* b_iou   = (const float*)d_in[5];
    const float* U_f     = (const float*)d_in[6];
    const float* b_f     = (const float*)d_in[7];
    float* out = (float*)d_out;

    // workspace layout
    char* ws = (char*)d_ws;
    int* counts = (int*)ws;                                              // 16 B (padded to 256)
    int* idx    = (int*)(ws + 256);                                      // T*N ints = 1 MB
    bf16_t* UfT   = (bf16_t*)(ws + 256 + (size_t)T_TYPES * N_NODES * 4); // 128 KB
    bf16_t* UiouT = (bf16_t*)((char*)UfT + (size_t)T_TYPES * H * H * 2); // 384 KB

    k_init<<<1, 32, 0, stream>>>(counts);
    int prepThreads = T_TYPES * H * H + T_TYPES * 3 * H * H;             // 262144
    k_prep<<<(prepThreads + 255) / 256, 256, 0, stream>>>(U_f, U_iou, UfT, UiouT);
    k_bucket<<<N_NODES / 256, 256, 0, stream>>>(type_id, counts, idx);
    k_compute<<<T_TYPES * TILES_PER_TYPE, 256, 0, stream>>>(
        h, c, f_input, b_iou, b_f, UfT, UiouT, counts, idx, out);
}
